// TRGLayer_3822520893602
// MI455X (gfx1250) — matrix operand order repaired and checked by reading, not compiled, not run on hardware
//
#include <hip/hip_runtime.h>
#include <hip/hip_bf16.h>

// ---------------- problem constants ----------------
constexpr int   Nn = 8, Cc = 256, HDh = 8, Ee = 64, Tt = 16, HSs = 14, WSs = 14;
constexpr int   Ss = 196, TS = 3136;            // S = 14*14, T*S
constexpr float EPSc   = 1e-5f;
constexpr float FACTOR = 1.0f / 112.0f;         // 1/sqrt(S*E) = 1/sqrt(12544)

// ---------------- workspace layout (bytes) ----------------
constexpr size_t HT_BYTES = (size_t)Nn * TS * Cc * 2;          // 12,845,056 each
constexpr size_t OFF_HT_T = 0;
constexpr size_t OFF_HT_P = OFF_HT_T + HT_BYTES;
constexpr size_t OFF_HT_G = OFF_HT_P + HT_BYTES;
constexpr size_t TH_BYTES = (size_t)Nn * 512 * TS * 2;         // 25,690,112
constexpr size_t OFF_TH   = OFF_HT_G + HT_BYTES;
constexpr size_t OFF_PH   = OFF_TH + TH_BYTES;
constexpr size_t GG_BYTES = (size_t)Nn * 2048 * TS * 2;        // 102,760,448
constexpr size_t OFF_GG   = OFF_PH + TH_BYTES;
constexpr size_t OFF_WT   = OFF_GG + GG_BYTES;                 // 512*256 f16
constexpr size_t OFF_WP   = OFF_WT + (size_t)512 * 256 * 2;
constexpr size_t OFF_WG   = OFF_WP + (size_t)512 * 256 * 2;    // 2048*256 f16
constexpr size_t OFF_BN1  = OFF_WG + (size_t)2048 * 256 * 2;   // 6*256 f32 (sc/sh x3)
constexpr size_t OFF_BN2  = OFF_BN1 + (size_t)6 * 256 * 4;     // 2*2048 f32
constexpr size_t OFF_ADJ  = OFF_BN2 + (size_t)2 * 2048 * 4;    // N*H*T*T f32
constexpr size_t OFF_HF   = OFF_ADJ + (size_t)Nn * HDh * Tt * Tt * 4;   // [n][t][h] f32
constexpr size_t OFF_ATTN = OFF_HF + (size_t)Nn * Tt * HDh * 4;         // [n][t][h] f32

// ---------------- types ----------------
typedef _Float16 v16h __attribute__((ext_vector_type(16)));
typedef _Float16 v8h  __attribute__((ext_vector_type(8)));
typedef float    v8f  __attribute__((ext_vector_type(8)));

union V16 { v16h v; v8h h2[2]; _Float16 e[16]; };
union V8F { v8f v; float e[8]; };

__device__ __forceinline__ float hswish_f(float x) {
    float t = fminf(fmaxf(x + 3.0f, 0.0f), 6.0f);
    return x * t * (1.0f / 6.0f);
}

// ================= K0: weight conversion + BN folding =================
__global__ __launch_bounds__(256) void prep_kernel(
    const float* tpw, const float* ppw, const float* gpw,
    const float* tg, const float* tb, const float* tm, const float* tv,
    const float* pg, const float* pb, const float* pm, const float* pv,
    const float* gg_, const float* gb, const float* gm, const float* gv,
    const float* g2g, const float* g2b, const float* g2m, const float* g2v,
    char* ws)
{
    int i = blockIdx.x * 256 + threadIdx.x;
    _Float16* wt = (_Float16*)(ws + OFF_WT);
    _Float16* wp = (_Float16*)(ws + OFF_WP);
    _Float16* wg = (_Float16*)(ws + OFF_WG);
    if (i < 512 * 256) { wt[i] = (_Float16)tpw[i]; wp[i] = (_Float16)ppw[i]; }
    if (i < 2048 * 256) { wg[i] = (_Float16)gpw[i]; }
    float* bn1 = (float*)(ws + OFF_BN1);
    if (i < 256) {
        float s0 = tg[i]  * rsqrtf(tv[i] + EPSc); bn1[i]        = s0; bn1[256 + i]  = tb[i] - tm[i] * s0;
        float s1 = pg[i]  * rsqrtf(pv[i] + EPSc); bn1[512 + i]  = s1; bn1[768 + i]  = pb[i] - pm[i] * s1;
        float s2 = gg_[i] * rsqrtf(gv[i] + EPSc); bn1[1024 + i] = s2; bn1[1280 + i] = gb[i] - gm[i] * s2;
    }
    float* bn2 = (float*)(ws + OFF_BN2);
    if (i < 2048) {
        float s = g2g[i] * rsqrtf(g2v[i] + EPSc);
        bn2[i] = s; bn2[2048 + i] = g2b[i] - g2m[i] * s;
    }
}

// ========== K1: fused 3-way depthwise conv(1,3,3) + BN1 + hswish -> f16 [n][ts][c] ==========
__global__ __launch_bounds__(256) void conv_kernel(
    const float* __restrict__ x, const float* __restrict__ tdw,
    const float* __restrict__ pdw, const float* __restrict__ gdw, char* ws)
{
    __shared__ float tile[32 * 197];       // 32 channels x 196, pitch 197 (conflict-free)
    const float* bn1 = (const float*)(ws + OFF_BN1);
    _Float16* hT = (_Float16*)(ws + OFF_HT_T);
    _Float16* hP = (_Float16*)(ws + OFF_HT_P);
    _Float16* hG = (_Float16*)(ws + OFF_HT_G);
    int n = blockIdx.x >> 4, t = blockIdx.x & 15;
    int tid = threadIdx.x, cl = tid & 31;
    for (int c0 = 0; c0 < 256; c0 += 32) {
        int c = c0 + cl;
        float w0[9], w1[9], w2[9];
        #pragma unroll
        for (int k = 0; k < 9; ++k) { w0[k] = tdw[c * 9 + k]; w1[k] = pdw[c * 9 + k]; w2[k] = gdw[c * 9 + k]; }
        float sct = bn1[c],        sht = bn1[256 + c];
        float scp = bn1[512 + c],  shp = bn1[768 + c];
        float scg = bn1[1024 + c], shg = bn1[1280 + c];
        __syncthreads();                       // previous tile fully consumed
        for (int idx = tid; idx < 32 * 196; idx += 256) {
            int ci = idx / 196, s = idx - ci * 196;
            tile[ci * 197 + s] = x[(((size_t)n * 256 + c0 + ci) * 16 + t) * 196 + s];
        }
        __syncthreads();
        for (int s = tid >> 5; s < 196; s += 8) {
            int hs = s / 14, wsx = s - hs * 14;
            float a0 = 0.f, a1 = 0.f, a2 = 0.f;
            #pragma unroll
            for (int dh = -1; dh <= 1; ++dh) {
                int ih = hs + dh; if (ih < 0 || ih > 13) continue;
                #pragma unroll
                for (int dw = -1; dw <= 1; ++dw) {
                    int iw = wsx + dw; if (iw < 0 || iw > 13) continue;
                    float v = tile[cl * 197 + ih * 14 + iw];
                    int k = (dh + 1) * 3 + (dw + 1);
                    a0 += v * w0[k]; a1 += v * w1[k]; a2 += v * w2[k];
                }
            }
            size_t o = ((size_t)n * TS + t * 196 + s) * 256 + c;   // [n][ts][c]
            hT[o] = (_Float16)hswish_f(a0 * sct + sht);
            hP[o] = (_Float16)hswish_f(a1 * scp + shp);
            hG[o] = (_Float16)hswish_f(a2 * scg + shg);
        }
    }
}

// ========== K2: WMMA GEMM (2x2 register tile, 32x32 per wave) ==========
// Y[n][o][ts] = sum_c W[o][c] * hT[n][ts][c] (+epilogue)
// mode 0: +bias[o]; mode 1: bn2 scale/shift then hswish
__global__ __launch_bounds__(256) void gemm_kernel(
    const _Float16* __restrict__ W, const _Float16* __restrict__ B,
    _Float16* __restrict__ Y, int O,
    const float* __restrict__ bias, const float* __restrict__ sc2,
    const float* __restrict__ sh2, int mode)
{
    int wave = (blockIdx.x << 3) + (threadIdx.x >> 5);
    int lane = threadIdx.x & 31;
    int tilesO2 = O >> 5;               // 32-row macro tiles
    int perN = tilesO2 * 98;            // 98 = 196/2 macro j-tiles
    int n = wave / perN;
    int r = wave - n * perN;
    int ot = r / 98, jt = r - (r / 98) * 98;
    int halfI = lane >> 4, l15 = lane & 15;
    int obase = ot * 32, jbase = jt * 32;
    const _Float16* arow0 = W + (size_t)(obase + l15) * 256;
    const _Float16* arow1 = arow0 + (size_t)16 * 256;
    const _Float16* brow0 = B + ((size_t)n * TS + jbase + l15) * 256;
    const _Float16* brow1 = brow0 + (size_t)16 * 256;
    v8f acc00 = {0.f,0.f,0.f,0.f,0.f,0.f,0.f,0.f};
    v8f acc01 = acc00, acc10 = acc00, acc11 = acc00;
    #pragma unroll 2
    for (int k0 = 0; k0 < 256; k0 += 32) {
        V16 a0, a1, b0, b1;
        a0.h2[0] = *(const v8h*)(arow0 + k0 + halfI * 8);
        a0.h2[1] = *(const v8h*)(arow0 + k0 + 16 + halfI * 8);
        a1.h2[0] = *(const v8h*)(arow1 + k0 + halfI * 8);
        a1.h2[1] = *(const v8h*)(arow1 + k0 + 16 + halfI * 8);
        b0.h2[0] = *(const v8h*)(brow0 + k0 + halfI * 8);
        b0.h2[1] = *(const v8h*)(brow0 + k0 + 16 + halfI * 8);
        b1.h2[0] = *(const v8h*)(brow1 + k0 + halfI * 8);
        b1.h2[1] = *(const v8h*)(brow1 + k0 + 16 + halfI * 8);
        acc00 = __builtin_amdgcn_wmma_f32_16x16x32_f16(false, a0.v, false, b0.v,
                                                       (short)0, acc00, false, false);
        acc01 = __builtin_amdgcn_wmma_f32_16x16x32_f16(false, a0.v, false, b1.v,
                                                       (short)0, acc01, false, false);
        acc10 = __builtin_amdgcn_wmma_f32_16x16x32_f16(false, a1.v, false, b0.v,
                                                       (short)0, acc10, false, false);
        acc11 = __builtin_amdgcn_wmma_f32_16x16x32_f16(false, a1.v, false, b1.v,
                                                       (short)0, acc11, false, false);
    }
    V8F d[4];
    d[0].v = acc00; d[1].v = acc01; d[2].v = acc10; d[3].v = acc11;
    #pragma unroll
    for (int io = 0; io < 2; ++io) {
        #pragma unroll
        for (int ij = 0; ij < 2; ++ij) {
            V8F& dd = d[io * 2 + ij];
            #pragma unroll
            for (int v = 0; v < 8; ++v) {
                int o = obase + io * 16 + v + (halfI << 3);
                int j = jbase + ij * 16 + l15;
                float val = dd.e[v];
                if (mode == 0) val += bias[o];
                else           val  = hswish_f(val * sc2[o] + sh2[o]);
                Y[((size_t)n * O + o) * TS + j] = (_Float16)val;
            }
        }
    }
}

// ========== K3: WMMA adjacency  adj[n,h,t,u] = softmax_u( F * sum_{e,s} TH*PH ) ==========
__global__ __launch_bounds__(32) void adj_kernel(char* ws)
{
    __shared__ float lt[16 * 17];
    const _Float16* th = (const _Float16*)(ws + OFF_TH);
    const _Float16* ph = (const _Float16*)(ws + OFF_PH);
    float* adj = (float*)(ws + OFF_ADJ);
    int n = blockIdx.x >> 3, h = blockIdx.x & 7;
    int lane = threadIdx.x & 31;
    int halfI = lane >> 4, l15 = lane & 15;
    const _Float16* A0 = th + ((size_t)n * 512 + h * 64) * TS + l15 * 196;  // row t=l15
    const _Float16* B0 = ph + ((size_t)n * 512 + h * 64) * TS + l15 * 196;  // col u=l15
    v8f acc = {0.f, 0.f, 0.f, 0.f, 0.f, 0.f, 0.f, 0.f};
    for (int k0 = 0; k0 < 64 * 196; k0 += 32) {          // K = E*S = 12544
        V16 a, b;
        #pragma unroll
        for (int i = 0; i < 16; ++i) {
            int ka = k0 + ((i < 8) ? 0 : 16) + halfI * 8 + (i & 7);
            int ea = ka / 196, sa = ka - ea * 196;
            a.e[i] = A0[(size_t)ea * TS + sa];
            int kb = k0 + ((i < 8) ? 0 : 16) + halfI * 8 + (i & 7);
            int eb = kb / 196, sb = kb - eb * 196;
            b.e[i] = B0[(size_t)eb * TS + sb];
        }
        acc = __builtin_amdgcn_wmma_f32_16x16x32_f16(false, a.v, false, b.v,
                                                     (short)0, acc, false, false);
    }
    V8F d; d.v = acc;
    #pragma unroll
    for (int v = 0; v < 8; ++v) lt[(v + halfI * 8) * 17 + l15] = FACTOR * d.e[v];
    __syncthreads();
    if (lane < 16) {
        int t = lane;
        float mx = -1e30f;
        for (int u = 0; u < 16; ++u) mx = fmaxf(mx, lt[t * 17 + u]);
        float ex[16], sum = 0.f;
        for (int u = 0; u < 16; ++u) { ex[u] = __expf(lt[t * 17 + u] - mx); sum += ex[u]; }
        float inv = 1.f / sum;
        for (int u = 0; u < 16; ++u)
            adj[(((size_t)n * 8 + h) * 16 + t) * 16 + u] = ex[u] * inv;
    }
}

// ========== K4: head features  hf[n][t][h] = mean_{c,s} hswish( adj . G5 ) ==========
__global__ __launch_bounds__(256) void hf_kernel(char* ws)
{
    __shared__ float adjL[256];
    __shared__ float red[256 * 16];
    const _Float16* gg = (const _Float16*)(ws + OFF_GG);
    const float* adj = (const float*)(ws + OFF_ADJ);
    float* hf = (float*)(ws + OFF_HF);
    int n = blockIdx.x >> 3, h = blockIdx.x & 7;
    int tid = threadIdx.x;
    adjL[tid] = adj[((size_t)n * 8 + h) * 256 + tid];
    __syncthreads();
    float part[16];
    #pragma unroll
    for (int t = 0; t < 16; ++t) part[t] = 0.f;
    const _Float16* g0 = gg + ((size_t)n * 2048 + h * 256) * TS;
    for (int item = tid; item < 256 * 196; item += 256) {
        int c = item / 196, s = item - c * 196;
        const _Float16* gp = g0 + (size_t)c * TS + s;
        float g5[16];
        #pragma unroll
        for (int u = 0; u < 16; ++u) g5[u] = (float)gp[u * 196];
        #pragma unroll
        for (int t = 0; t < 16; ++t) {
            float a = 0.f;
            #pragma unroll
            for (int u = 0; u < 16; ++u) a += adjL[t * 16 + u] * g5[u];
            part[t] += hswish_f(a);
        }
    }
    #pragma unroll
    for (int t = 0; t < 16; ++t) red[tid * 16 + t] = part[t];
    __syncthreads();
    if (tid < 16) {
        float s = 0.f;
        for (int i = 0; i < 256; ++i) s += red[i * 16 + tid];
        hf[((size_t)n * 16 + tid) * 8 + h] = s * (1.0f / (256.0f * 196.0f));
    }
}

// ========== K5: head-mix attention  attn[n][t][:] = softmax(hf[n][t][:] @ heads_w) ==========
__global__ void attn_kernel(const float* __restrict__ heads_w, char* ws)
{
    int i = blockIdx.x * blockDim.x + threadIdx.x;     // i = n*16 + t
    if (i >= Nn * Tt) return;
    const float* hf = (const float*)(ws + OFF_HF);
    float* attn = (float*)(ws + OFF_ATTN);
    float hv[8], lg[8];
    #pragma unroll
    for (int h = 0; h < 8; ++h) hv[h] = hf[i * 8 + h];
    float mx = -1e30f;
    #pragma unroll
    for (int j = 0; j < 8; ++j) {
        float a = 0.f;
        #pragma unroll
        for (int h = 0; h < 8; ++h) a += hv[h] * heads_w[h * 8 + j];
        lg[j] = a; mx = fmaxf(mx, a);
    }
    float sum = 0.f;
    #pragma unroll
    for (int j = 0; j < 8; ++j) { lg[j] = __expf(lg[j] - mx); sum += lg[j]; }
    float inv = 1.f / sum;
    #pragma unroll
    for (int j = 0; j < 8; ++j) attn[i * 8 + j] = lg[j] * inv;
}

// ========== K6: recompute gcn, combine heads, residual + hswish ==========
__global__ __launch_bounds__(256) void out_kernel(const float* __restrict__ x,
                                                  char* ws, float* __restrict__ out)
{
    __shared__ float adjL[2048];
    __shared__ float attnL[128];
    const _Float16* gg = (const _Float16*)(ws + OFF_GG);
    const float* adj = (const float*)(ws + OFF_ADJ);
    const float* attn = (const float*)(ws + OFF_ATTN);
    int n = blockIdx.x >> 8, c = blockIdx.x & 255;
    int tid = threadIdx.x;
    for (int i = tid; i < 2048; i += 256) adjL[i] = adj[(size_t)n * 2048 + i];
    if (tid < 128) attnL[tid] = attn[(size_t)n * 128 + tid];
    __syncthreads();
    if (tid < 196) {
        int s = tid;
        float z[16];
        #pragma unroll
        for (int t = 0; t < 16; ++t) z[t] = 0.f;
        #pragma unroll 1
        for (int h = 0; h < 8; ++h) {
            const _Float16* gp = gg + ((size_t)n * 2048 + h * 256 + c) * TS + s;
            float g5[16];
            #pragma unroll
            for (int u = 0; u < 16; ++u) g5[u] = (float)gp[u * 196];
            #pragma unroll
            for (int t = 0; t < 16; ++t) {
                float a = 0.f;
                #pragma unroll
                for (int u = 0; u < 16; ++u) a += adjL[(h * 16 + t) * 16 + u] * g5[u];
                z[t] += attnL[t * 8 + h] * hswish_f(a);
            }
        }
        size_t base = ((size_t)n * 256 + c) * TS + s;
        #pragma unroll
        for (int t = 0; t < 16; ++t) {
            float xv = x[base + t * 196];
            out[base + t * 196] = hswish_f(xv + z[t]);
        }
    }
}

// ---------------- host launch ----------------
extern "C" void kernel_launch(void* const* d_in, const int* in_sizes, int n_in,
                              void* d_out, int out_size, void* d_ws, size_t ws_size,
                              hipStream_t stream) {
    (void)in_sizes; (void)n_in; (void)out_size; (void)ws_size;
    const float* x    = (const float*)d_in[0];
    const float* tdw  = (const float*)d_in[1];
    const float* tg   = (const float*)d_in[2];
    const float* tb   = (const float*)d_in[3];
    const float* tm   = (const float*)d_in[4];
    const float* tv   = (const float*)d_in[5];
    const float* tpw  = (const float*)d_in[6];
    const float* pdw  = (const float*)d_in[7];
    const float* pg   = (const float*)d_in[8];
    const float* pb   = (const float*)d_in[9];
    const float* pm   = (const float*)d_in[10];
    const float* pv   = (const float*)d_in[11];
    const float* ppw  = (const float*)d_in[12];
    const float* gdw  = (const float*)d_in[13];
    const float* gg1  = (const float*)d_in[14];
    const float* gb1  = (const float*)d_in[15];
    const float* gm1  = (const float*)d_in[16];
    const float* gv1  = (const float*)d_in[17];
    const float* gpw  = (const float*)d_in[18];
    const float* tpb  = (const float*)d_in[19];
    const float* ppb  = (const float*)d_in[20];
    const float* g2g  = (const float*)d_in[21];
    const float* g2b  = (const float*)d_in[22];
    const float* g2m  = (const float*)d_in[23];
    const float* g2v  = (const float*)d_in[24];
    const float* hw   = (const float*)d_in[25];
    char* ws = (char*)d_ws;
    float* out = (float*)d_out;

    prep_kernel<<<2048, 256, 0, stream>>>(tpw, ppw, gpw,
        tg, tb, tm, tv, pg, pb, pm, pv, gg1, gb1, gm1, gv1,
        g2g, g2b, g2m, g2v, ws);

    conv_kernel<<<Nn * Tt, 256, 0, stream>>>(x, tdw, pdw, gdw, ws);

    const _Float16* wsT = (const _Float16*)(ws + OFF_WT);
    const _Float16* wsP = (const _Float16*)(ws + OFF_WP);
    const _Float16* wsG = (const _Float16*)(ws + OFF_WG);
    const _Float16* hTT = (const _Float16*)(ws + OFF_HT_T);
    const _Float16* hTP = (const _Float16*)(ws + OFF_HT_P);
    const _Float16* hTG = (const _Float16*)(ws + OFF_HT_G);
    const float* bn2sc = (const float*)(ws + OFF_BN2);
    const float* bn2sh = bn2sc + 2048;

    // theta/phi: 8n * 16 macro-o-tiles * 98 macro-j-tiles = 12544 waves / 8 per block
    gemm_kernel<<<1568, 256, 0, stream>>>(wsT, hTT, (_Float16*)(ws + OFF_TH),
                                          512, tpb, nullptr, nullptr, 0);
    gemm_kernel<<<1568, 256, 0, stream>>>(wsP, hTP, (_Float16*)(ws + OFF_PH),
                                          512, ppb, nullptr, nullptr, 0);
    // g: 8n * 64 macro-o-tiles * 98 macro-j-tiles = 50176 waves / 8 per block
    gemm_kernel<<<6272, 256, 0, stream>>>(wsG, hTG, (_Float16*)(ws + OFF_GG),
                                          2048, nullptr, bn2sc, bn2sh, 1);

    adj_kernel<<<Nn * HDh, 32, 0, stream>>>(ws);
    hf_kernel<<<Nn * HDh, 256, 0, stream>>>(ws);
    attn_kernel<<<1, 128, 0, stream>>>(hw, ws);
    out_kernel<<<Nn * Cc, 256, 0, stream>>>(x, ws, out);
}
